// SwapV2_18957985644705
// MI455X (gfx1250) — matrix operand order repaired and checked by reading, not compiled, not run on hardware
//
#include <hip/hip_runtime.h>
#include <hip/hip_fp16.h>

typedef __attribute__((ext_vector_type(16))) _Float16 v16h;
typedef __attribute__((ext_vector_type(8)))  _Float16 v8h;
typedef __attribute__((ext_vector_type(8)))  float    v8f;
typedef __attribute__((ext_vector_type(2)))  float    v2f;

#define Bdim 8
#define Cdim 32
#define Hdim 64
#define Wdim 64
#define WAVES 8
#define TPIX  (16 * WAVES)                    // 128 output pixels per block
#define NBLK  (Bdim * Hdim * Wdim / TPIX)     // 256 blocks

// ---------------------------------------------------------------------------
// Kernel 1: f32 -> f16 swizzle of the input into WMMA *A*-fragment order.
// A tile (c, ut, kt) is X[u = ut*16 + lo][v = kt*32 + kmap(e,hi)], 16x32 f16.
// kmap(e,hi) = e + 16*hi  -> per-lane v16h is one contiguous
// 16-float run of the source row, so this kernel does contiguous 32B reads.
// v16h index: (((b*C + c)*4 + ut)*2 + kt)*32 + lane
// ---------------------------------------------------------------------------
__global__ __launch_bounds__(256)
void swizzle_a_frags(const float* __restrict__ inp, _Float16* __restrict__ F)
{
    const long t  = (long)blockIdx.x * blockDim.x + threadIdx.x;   // 131,072
    const long d0 = t * 8;                     // first dest half-index
    const int e0   = (int)(d0 & 15);           // 0 or 8
    const int lane = (int)((d0 >> 4) & 31);
    const int lo = lane & 15, hi = lane >> 4;
    const int kt = (int)((d0 >> 9) & 1);
    const int ut = (int)((d0 >> 10) & 3);
    const int c  = (int)((d0 >> 12) & 31);
    const int b  = (int)(d0 >> 17);

    const int u  = ut * 16 + lo;
    const int vb = kt * 32 + 16 * hi + (e0 ? 8 : 0);
    const float* __restrict__ X = inp + ((size_t)(b * Cdim + c) * Hdim) * Wdim;

    v8h pk;
    #pragma unroll
    for (int i = 0; i < 8; ++i) pk[i] = (_Float16)X[u * Wdim + vb + i];
    *reinterpret_cast<v8h*>(F + d0) = pk;
}

// ---------------------------------------------------------------------------
// Kernel 2: block = 8 waves; each wave owns one 16-pixel tile and loops over
// all 32 channels. Per channel: preload all 8 A-fragments (16-deep b128 load
// clause, staggered waits), G = X(64x64) * Wx^T(64x16) via 8 WMMA, then
// out[p] = sum_u wy[p,u] * G[u,p] as 16 packed-f32 FMAs over 4 independent
// chains + one half-swap shuffle + one coalesced store.
// ---------------------------------------------------------------------------
__global__ __launch_bounds__(32 * WAVES)
void gauss_warp_wmma(const v16h* __restrict__ Fv,
                     const float* __restrict__ exPx,
                     const float* __restrict__ exPy,
                     const float* __restrict__ sigmax,
                     const float* __restrict__ sigmay,
                     float* __restrict__ out)
{
    __shared__ float    wyS[TPIX][Hdim];   // row weights, f32 (epilogue)  32 KB
    __shared__ _Float16 wxS[TPIX][Wdim];   // col weights, f16 (WMMA B)    16 KB
    __shared__ float    zinvS[TPIX];

    const int b   = blockIdx.x / (Hdim * Wdim / TPIX);
    const int p0  = (blockIdx.x % (Hdim * Wdim / TPIX)) * TPIX;
    const int tid = threadIdx.x;

    // ---- Stage 1: separable Gaussian weights + normalization ----
    for (int idx = tid; idx < TPIX * Hdim; idx += blockDim.x) {
        const int p = idx >> 6, u = idx & 63;
        const int gp = b * Hdim * Wdim + p0 + p;
        const float d = ((float)u - exPy[gp]) / sigmay[gp];
        wyS[p][u] = __expf(-0.5f * d * d);
    }
    for (int idx = tid; idx < TPIX * Wdim; idx += blockDim.x) {
        const int p = idx >> 6, v = idx & 63;
        const int gp = b * Hdim * Wdim + p0 + p;
        const float d = ((float)v - exPx[gp]) / sigmax[gp];
        wxS[p][v] = (_Float16)__expf(-0.5f * d * d);
    }
    __syncthreads();
    if (tid < TPIX) {
        float sy = 0.f, sx = 0.f;
        #pragma unroll 8
        for (int u = 0; u < Hdim; ++u) sy += wyS[tid][u];
        #pragma unroll 8
        for (int v = 0; v < Wdim; ++v) sx += (float)wxS[tid][v];
        zinvS[tid] = 1.0f / (sy * sx + 1e-8f);
    }
    __syncthreads();

    const int wave = tid >> 5, lane = tid & 31;
    const int lo = lane & 15, hi = lane >> 4;
    const int prow = wave * 16 + lo;         // this lane's output pixel (in block)

    // B fragments (Wx^T: K=v, N=p), kept in registers for all channels.
    // B k order used here (same as A): lane N=lo; element e -> K = e + 16*hi (within 32-K step)
    v16h bfrag[2];
    #pragma unroll
    for (int kt = 0; kt < 2; ++kt)
        #pragma unroll
        for (int e = 0; e < 16; ++e)
            bfrag[kt][e] = wxS[prow][kt * 32 + e + 16 * hi];

    // wy for the epilogue, as adjacent pairs: u = ut*16 + (2j, 2j+1) + 8*hi
    v2f wyr2[4][4];
    #pragma unroll
    for (int ut = 0; ut < 4; ++ut)
        #pragma unroll
        for (int j = 0; j < 4; ++j) {
            const int u = ut * 16 + 2 * j + 8 * hi;
            wyr2[ut][j] = v2f{wyS[prow][u], wyS[prow][u + 1]};
        }

    const float zi = zinvS[prow];
    const size_t outPix = (size_t)p0 + prow;

    for (int c = 0; c < Cdim; ++c) {
        const long fb = (long)(b * Cdim + c) * 256;   // v16h units

        // Preload all 8 A fragments into distinct registers: one deep load
        // clause, so s_wait_loadcnt can stagger instead of draining to 0.
        v16h af[8];
        #pragma unroll
        for (int i = 0; i < 8; ++i)            // i = ut*2 + kt
            af[i] = Fv[fb + i * 32 + lane];    // coalesced 32B/lane

        v8f acc[4] = {};
        #pragma unroll
        for (int kt = 0; kt < 2; ++kt)
            #pragma unroll
            for (int ut = 0; ut < 4; ++ut)
                acc[ut] = __builtin_amdgcn_wmma_f32_16x16x32_f16(
                    false, af[ut * 2 + kt], false, bfrag[kt],
                    (short)0, acc[ut], false, false);

        // out[p=lo] = sum_u wy[p,u] * G[u,p]; lane holds rows r+8*hi of each ut.
        // 4 independent packed-f32 accumulator chains -> v_pk_fma_f32.
        v2f sp0 = {0.f, 0.f}, sp1 = {0.f, 0.f}, sp2 = {0.f, 0.f}, sp3 = {0.f, 0.f};
        #pragma unroll
        for (int ut = 0; ut < 4; ++ut) {
            sp0 += v2f{acc[ut][0], acc[ut][1]} * wyr2[ut][0];
            sp1 += v2f{acc[ut][2], acc[ut][3]} * wyr2[ut][1];
            sp2 += v2f{acc[ut][4], acc[ut][5]} * wyr2[ut][2];
            sp3 += v2f{acc[ut][6], acc[ut][7]} * wyr2[ut][3];
        }
        const v2f spA = sp0 + sp1, spB = sp2 + sp3;
        const v2f spT = spA + spB;
        float s = spT.x + spT.y;

        s += __shfl_xor(s, 16, 32);               // combine the two u-halves
        if (hi == 0)
            out[(size_t)(b * Cdim + c) * (Hdim * Wdim) + outPix] = s * zi;
    }
}

extern "C" void kernel_launch(void* const* d_in, const int* in_sizes, int n_in,
                              void* d_out, int out_size, void* d_ws, size_t ws_size,
                              hipStream_t stream) {
    const float* inp    = (const float*)d_in[0];
    const float* exPx   = (const float*)d_in[1];
    const float* exPy   = (const float*)d_in[2];
    const float* sigmax = (const float*)d_in[3];
    const float* sigmay = (const float*)d_in[4];
    float* out = (float*)d_out;

    _Float16* F = (_Float16*)d_ws;   // needs B*C*H*W*2 = 2 MiB of scratch

    const long nThreads = (long)Bdim * Cdim * Hdim * Wdim / 8;   // 131,072
    swizzle_a_frags<<<dim3((unsigned)(nThreads / 256)), dim3(256), 0, stream>>>(inp, F);

    gauss_warp_wmma<<<dim3(NBLK), dim3(32 * WAVES), 0, stream>>>(
        (const v16h*)F, exPx, exPy, sigmax, sigmay, out);
}
